// SVConv2d_66228395704654
// MI455X (gfx1250) — matrix operand order repaired and checked by reading, not compiled, not run on hardware
//
#include <hip/hip_runtime.h>

// ---------------------------------------------------------------------------
// SVConv2d (frequency-dependent 3x3 conv) for MI455X / gfx1250.
//
// Strategy: 161 independent GEMMs (one per output frequency f):
//   Out[(b,t), o] = sum_{m,n,i} X[b, i, t+m-1, f+n-1] * W[o, i, f, m, n]
// Computed with v_wmma_f32_16x16x32_bf16 using the bf16 hi/lo split
// (x = hi + lo; keep ah*bh + ah*bl + al*bh; ~1.5e-5 rel err, fp32 accum),
// which is ~8x the per-instruction MAC rate of the fp32 K=4 WMMA path and
// puts the kernel near the 23.3 TB/s memory roofline instead of 5x compute
// bound.
//
// Workspace requirement: ~193 MB in d_ws:
//   Xp_hi/Xp_lo : [B][T][F][CIN]    bf16   (84.4 MB each)
//   Wp_hi/Wp_lo : [F][3][3][CIN][COUT] bf16 (11.9 MB each)
// ---------------------------------------------------------------------------

#define B_    8
#define CIN_  64
#define COUT_ 64
#define T_    512
#define F_    161

typedef __attribute__((ext_vector_type(16))) __bf16 v16bf;
typedef __attribute__((ext_vector_type(8)))  __bf16 v8bf;
typedef __attribute__((ext_vector_type(8)))  float  v8f;

// ---------------------------------------------------------------------------
// Pass 1: x (B,CIN,T,F) f32 -> Xp (B,T,F,CIN) bf16 hi/lo, i contiguous.
// LDS tile transpose so both global read (along f) and global write (along i)
// are coalesced.
// ---------------------------------------------------------------------------
__global__ __launch_bounds__(256) void repack_x_kernel(
    const float* __restrict__ x, __bf16* __restrict__ xh, __bf16* __restrict__ xl) {
  __shared__ __bf16 sh[64 * 65];
  __shared__ __bf16 sl[64 * 65];
  int blk = blockIdx.x;          // (b*T + t)*3 + ftile
  int ft  = blk % 3;
  int bt  = blk / 3;             // b*T + t
  int t   = bt % T_;
  int b   = bt / T_;
  int f0  = ft * 64;
  int tid = threadIdx.x;

  // read: 64 i-rows x 64 f-cols, coalesced along f
  for (int idx = tid; idx < 64 * 64; idx += 256) {
    int i  = idx >> 6;
    int fl = idx & 63;
    int f  = f0 + fl;
    float v = 0.0f;
    if (f < F_) v = x[(((size_t)b * CIN_ + i) * T_ + t) * F_ + f];
    __bf16 h = (__bf16)v;
    __bf16 l = (__bf16)(v - (float)h);
    sh[fl * 65 + i] = h;   // transpose on the way in
    sl[fl * 65 + i] = l;
  }
  __syncthreads();
  // write: coalesced along i
  for (int idx = tid; idx < 64 * 64; idx += 256) {
    int fl = idx >> 6;
    int i  = idx & 63;
    int f  = f0 + fl;
    if (f < F_) {
      size_t o = ((size_t)bt * F_ + f) * CIN_ + i;
      xh[o] = sh[fl * 65 + i];
      xl[o] = sl[fl * 65 + i];
    }
  }
}

// ---------------------------------------------------------------------------
// Pass 2: kernel (COUT,CIN,F,3,3) f32 -> Wp (F,3,3,CIN,COUT) bf16 hi/lo,
// o contiguous (coalesced writes; gather reads are only 5.9M elements).
// ---------------------------------------------------------------------------
__global__ __launch_bounds__(256) void repack_w_kernel(
    const float* __restrict__ k, __bf16* __restrict__ wh, __bf16* __restrict__ wl) {
  size_t idx = (size_t)blockIdx.x * 256 + threadIdx.x;
  const size_t total = (size_t)F_ * 9 * CIN_ * COUT_;
  if (idx >= total) return;
  int    o  = (int)(idx & 63);
  size_t r  = idx >> 6;
  int    i  = (int)(r & 63); r >>= 6;
  int    mn = (int)(r % 9);
  int    f  = (int)(r / 9);
  int    m  = mn / 3, n = mn % 3;
  float v = k[((((size_t)o * CIN_ + i) * F_ + f) * 3 + m) * 3 + n];
  __bf16 h = (__bf16)v;
  wh[idx] = h;
  wl[idx] = (__bf16)(v - (float)h);
}

// ---------------------------------------------------------------------------
// Main GEMM kernel. grid = (F, 32); block = 256 threads = 8 waves.
// Wave w: 16 t-rows x 64 o-cols of one (f, m-block) output tile.
// ---------------------------------------------------------------------------
__device__ inline v16bf zero_frag() {
  v16bf r;
#pragma unroll
  for (int e = 0; e < 16; ++e) r[e] = (__bf16)0.0f;
  return r;
}

// A fragment (16-bit A 16x32 layout): lane holds K={0..7,16..23} (lanes 0-15)
// or K={8..15,24..31} (lanes 16-31) as two 16B contiguous chunks.
__device__ inline v16bf load_a_frag(const __bf16* __restrict__ p) {
  v8bf p0 = *(const v8bf*)(p);
  v8bf p1 = *(const v8bf*)(p + 16);
  v16bf r;
#pragma unroll
  for (int e = 0; e < 8; ++e) { r[e] = p0[e]; r[e + 8] = p1[e]; }
  return r;
}

__global__ __launch_bounds__(256) void svconv_wmma_kernel(
    const __bf16* __restrict__ xh, const __bf16* __restrict__ xl,
    const __bf16* __restrict__ wh, const __bf16* __restrict__ wl,
    const float* __restrict__ bias, float* __restrict__ out) {
  const int f    = blockIdx.x;        // frequency, 0..160
  const int mb   = blockIdx.y;        // 0..31 -> b = mb/4, t block = (mb%4)*128
  const int b    = mb >> 2;
  const int tid  = threadIdx.x;
  const int wave = tid >> 5;
  const int lane = tid & 31;
  const int lm   = lane & 15;
  const bool hi_half = lane >= 16;
  const int t_base = ((mb & 3) * 128) + wave * 16;
  const int t_row  = t_base + lm;     // this lane's A row (M = lane%16)

  v8f acc[4];
#pragma unroll
  for (int nt = 0; nt < 4; ++nt) acc[nt] = (v8f){0, 0, 0, 0, 0, 0, 0, 0};

  for (int m = 0; m < 3; ++m) {
    const int tp = t_row + m - 1;
    const bool t_ok = (tp >= 0) && (tp < T_);
    for (int n = 0; n < 3; ++n) {
      const int fp = f + n - 1;
      const bool ok = t_ok && (fp >= 0) && (fp < F_);   // per-lane halo mask
#pragma unroll
      for (int c = 0; c < 2; ++c) {                     // CIN in chunks of 32
        // ---- A fragments (hi/lo), zero outside the halo ----
        v16bf a_h = zero_frag();
        v16bf a_l = zero_frag();
        if (ok) {
          size_t aoff = (((size_t)b * T_ + tp) * F_ + fp) * CIN_ + c * 32 + (hi_half ? 8 : 0);
          a_h = load_a_frag(xh + aoff);
          a_l = load_a_frag(xl + aoff);
        }
        // ---- B fragments: lane N = o (lane%16), elements = K (i) in A's table order ----
        const size_t wrow =
            (((size_t)f * 9 + m * 3 + n) * CIN_ + (c * 32)) * COUT_;
        const __bf16* wbh = wh + wrow;
        const __bf16* wbl = wl + wrow;
#pragma unroll
        for (int nt = 0; nt < 4; ++nt) {
          v16bf b_h, b_l;
#pragma unroll
          for (int e = 0; e < 16; ++e) {
            const int kk = (e < 8 ? e : e + 8) + (hi_half ? 8 : 0);
            const size_t wo = (size_t)kk * COUT_ + nt * 16 + lm;
            b_h[e] = wbh[wo];
            b_l[e] = wbl[wo];
          }
          // bf16x3: hi*hi + hi*lo + lo*hi, fp32 accumulate
          acc[nt] = __builtin_amdgcn_wmma_f32_16x16x32_bf16(
              false, a_h, false, b_h, (short)0, acc[nt], false, false);
          acc[nt] = __builtin_amdgcn_wmma_f32_16x16x32_bf16(
              false, a_h, false, b_l, (short)0, acc[nt], false, false);
          acc[nt] = __builtin_amdgcn_wmma_f32_16x16x32_bf16(
              false, a_l, false, b_h, (short)0, acc[nt], false, false);
        }
      }
    }
  }

  // Epilogue: C/D layout -> lane N = lane%16, VGPR r -> M = r + 8*(lane>=16)
#pragma unroll
  for (int nt = 0; nt < 4; ++nt) {
    const int o = nt * 16 + lm;
    const float bv = bias[(size_t)o * F_ + f];
#pragma unroll
    for (int r = 0; r < 8; ++r) {
      const int t = t_base + r + (hi_half ? 8 : 0);
      out[(((size_t)b * COUT_ + o) * T_ + t) * F_ + f] = acc[nt][r] + bv;
    }
  }
}

// ---------------------------------------------------------------------------
extern "C" void kernel_launch(void* const* d_in, const int* in_sizes, int n_in,
                              void* d_out, int out_size, void* d_ws, size_t ws_size,
                              hipStream_t stream) {
  const float* x    = (const float*)d_in[0];
  const float* k    = (const float*)d_in[1];
  const float* bias = (const float*)d_in[2];
  float* out        = (float*)d_out;

  const size_t xsz = (size_t)B_ * T_ * F_ * CIN_;      // 42,205,184 elems
  const size_t wsz = (size_t)F_ * 9 * CIN_ * COUT_;    //  5,935,104 elems

  __bf16* xh = (__bf16*)d_ws;
  __bf16* xl = xh + xsz;
  __bf16* wh = xl + xsz;
  __bf16* wl = wh + wsz;   // total ws: (2*xsz + 2*wsz)*2 B ~= 193 MB

  repack_x_kernel<<<dim3(B_ * T_ * 3), 256, 0, stream>>>(x, xh, xl);
  repack_w_kernel<<<dim3((unsigned)((wsz + 255) / 256)), 256, 0, stream>>>(k, wh, wl);
  svconv_wmma_kernel<<<dim3(F_, 32), 256, 0, stream>>>(xh, xl, wh, wl, bias, out);
}
